// Feature_forcaster_4483945857239
// MI455X (gfx1250) — matrix operand order repaired and checked by reading, not compiled, not run on hardware
//
#include <hip/hip_runtime.h>
#include <hip/hip_bf16.h>
#include <math.h>

typedef __bf16 bf16;
typedef __attribute__((ext_vector_type(16))) bf16  bf16x16;
typedef __attribute__((ext_vector_type(8)))  float floatx8;

#define LSEQ 4096
#define EDIM 256
#define NBATCH 4
#define NHEADS 8

__device__ __forceinline__ floatx8 f8zero() {
  const floatx8 z = {0.f, 0.f, 0.f, 0.f, 0.f, 0.f, 0.f, 0.f};
  return z;
}

// ---------------------------------------------------------------------------
// WMMA fragment loaders, layouts per CDNA5 ISA 7.12.2 (wave32).
// A (16x32 bf16): lane L -> row M=L%16, half=L>=16; e<8 : K=8*half+e,
//                 e>=8 : K=16+8*half+(e-8)  -> two 16B contiguous chunks.
// B (32x16 bf16): lane L -> col N=L%16, half; same K mapping as A
//                 -> two 16B contiguous chunks.
// C/D layout: VGPR r -> row M=r+8*half, col N=lane%16.
// ---------------------------------------------------------------------------
__device__ __forceinline__ bf16x16 load_frag_a(const bf16* arow, int ld, int k0) {
  const int lane = threadIdx.x & 31;
  const int hf   = lane >> 4;
  const bf16* p = arow + (size_t)(lane & 15) * ld + k0 + 8 * hf;
  union { bf16x16 v; int4 q[2]; } u;
  u.q[0] = *reinterpret_cast<const int4*>(p);
  u.q[1] = *reinterpret_cast<const int4*>(p + 16);
  return u.v;
}

__device__ __forceinline__ bf16x16 load_frag_b(const bf16* brow, int ld, int k0) {
  const int lane = threadIdx.x & 31;
  const int hf   = lane >> 4;
  const bf16* p = brow + (size_t)(lane & 15) * ld + k0 + 8 * hf;
  union { bf16x16 v; int4 q[2]; } u;
  u.q[0] = *reinterpret_cast<const int4*>(p);
  u.q[1] = *reinterpret_cast<const int4*>(p + 16);
  return u.v;
}

// ---------------------------------------------------------------------------
// Generic WMMA GEMM:  C[m, o] = epi( sum_k A[m,k] * Bw[o,k] + bias[o] )
// OUTMODE 0: bf16 store (ldc)
// OUTMODE 2: f32 scatter, m = n*4096+l  ->  out[(n*256+o)*4096 + l]
// ---------------------------------------------------------------------------
template <int NT, int OUTMODE>
__global__ void gemm_wmma_kernel(const bf16* __restrict__ A, int lda,
                                 const bf16* __restrict__ Bw, int ldb,
                                 void* __restrict__ Cout, int ldc,
                                 const float* __restrict__ bias,
                                 int Kdim, float scale, int do_relu) {
  const int wave = threadIdx.x >> 5;
  const int lane = threadIdx.x & 31;
  const int hf   = lane >> 4;
  const int row0 = blockIdx.x * 128 + wave * 16;
  const int col0 = blockIdx.y * (NT * 16);
  const bf16* arow = A + (size_t)row0 * lda;

  floatx8 acc[NT];
#pragma unroll
  for (int t = 0; t < NT; ++t) acc[t] = f8zero();

  for (int k0 = 0; k0 < Kdim; k0 += 32) {
    if (k0 + 32 < Kdim)  // global_prefetch_b8
      __builtin_prefetch(arow + (size_t)(lane & 15) * lda + k0 + 32, 0, 1);
    bf16x16 a = load_frag_a(arow, lda, k0);
#pragma unroll
    for (int t = 0; t < NT; ++t) {
      bf16x16 b = load_frag_b(Bw + (size_t)(col0 + 16 * t) * ldb, ldb, k0);
      acc[t] = __builtin_amdgcn_wmma_f32_16x16x32_bf16(
          false, a, false, b, (short)0, acc[t], false, false);
    }
  }

  (void)ldc;
#pragma unroll
  for (int t = 0; t < NT; ++t) {
    const int col = col0 + 16 * t + (lane & 15);
    const float bv = bias ? bias[col] : 0.f;
#pragma unroll
    for (int r = 0; r < 8; ++r) {
      const int m = row0 + r + 8 * hf;
      float v = acc[t][r] + bv;
      if (do_relu) v = fmaxf(v, 0.f);
      v *= scale;
      if (OUTMODE == 0) {
        ((bf16*)Cout)[(size_t)m * ldc + col] = (bf16)v;
      } else {
        const int n = m >> 12, l = m & 4095;
        ((float*)Cout)[(((size_t)n * EDIM + col) << 12) + l] = v;
      }
    }
  }
}

// ---------------------------------------------------------------------------
// Fused multi-head attention for one (n, 16-query block).
//  - dynamic LDS: float amap_acc[16][4096] (256 KB) -- head-mean attention map
//  - per head, three WMMA sweeps over keys (QK^T recompute is free; exp isn't):
//      A: row max only          (0 exp/elem)
//      B: sum of exp(s-max)     (1 exp/elem)
//      C: P = exp(s-max)/sum -> amap (LDS) + staged P.V WMMA  (1 exp/elem)
//  - O written once per head, amap written once at the end.
// grid: (256, NBATCH), block: 256 threads = 8 waves; wave w owns keys
// [w*512, (w+1)*512).
// ---------------------------------------------------------------------------
__global__ void fused_attn_kernel(const bf16* __restrict__ Qp,
                                  const bf16* __restrict__ Kp,
                                  const bf16* __restrict__ Vt,  // [N][256][4096]
                                  bf16* __restrict__ Ob,
                                  float* __restrict__ amap) {
  const int n    = blockIdx.y;
  const int q0   = blockIdx.x * 16;
  const int tid  = threadIdx.x;
  const int wave = tid >> 5;
  const int lane = tid & 31;
  const int hf   = lane >> 4;
  const int lcol = lane & 15;

  extern __shared__ char smem_raw[];
  float* amap_acc = (float*)smem_raw;  // [16][LSEQ] = 256 KB
  __shared__ float s_pm[8][16];
  __shared__ float s_ps[8][16];
  __shared__ float s_rm[16];
  __shared__ float s_rs[16];
  __shared__ float s_ored[8][16][32];              // 16 KB
  __shared__ __align__(16) bf16 s_pst[8][16][32];  // 8 KB

  for (int i = tid; i < 16 * LSEQ; i += 256) amap_acc[i] = 0.f;
  __syncthreads();

  const bf16* Vtn = Vt + (size_t)n * EDIM * LSEQ;
  const size_t qkrow = ((size_t)n * LSEQ + q0) * EDIM;
  const int key0 = wave * 512;

  for (int h = 0; h < NHEADS; ++h) {
    const bf16x16 afrag = load_frag_a(Qp + qkrow + h * 32, EDIM, 0);
    const bf16* Kh = Kp + (size_t)n * LSEQ * EDIM + h * 32;

    // ---- pass A: row max over this wave's 512 keys (fmax only) ----
    float rmax[8];
#pragma unroll
    for (int r = 0; r < 8; ++r) rmax[r] = -INFINITY;
    for (int kt = 0; kt < 32; ++kt) {
      bf16x16 b = load_frag_b(Kh + (size_t)(key0 + kt * 16) * EDIM, EDIM, 0);
      floatx8 sacc = f8zero();
      sacc = __builtin_amdgcn_wmma_f32_16x16x32_bf16(
          false, afrag, false, b, (short)0, sacc, false, false);
#pragma unroll
      for (int r = 0; r < 8; ++r) rmax[r] = fmaxf(rmax[r], sacc[r]);
    }
#pragma unroll
    for (int off = 1; off < 16; off <<= 1)
#pragma unroll
      for (int r = 0; r < 8; ++r)
        rmax[r] = fmaxf(rmax[r], __shfl_xor(rmax[r], off, 32));
    if (lcol == 0)
#pragma unroll
      for (int r = 0; r < 8; ++r) s_pm[wave][hf * 8 + r] = rmax[r];
    __syncthreads();
    if (tid < 16) {
      float m = s_pm[0][tid];
      for (int w2 = 1; w2 < 8; ++w2) m = fmaxf(m, s_pm[w2][tid]);
      s_rm[tid] = m;
    }
    __syncthreads();
    float frm[8];
#pragma unroll
    for (int r = 0; r < 8; ++r) frm[r] = s_rm[hf * 8 + r];

    // ---- pass B: sum of exp(s - max) ----
    float rsum[8];
#pragma unroll
    for (int r = 0; r < 8; ++r) rsum[r] = 0.f;
    for (int kt = 0; kt < 32; ++kt) {
      bf16x16 b = load_frag_b(Kh + (size_t)(key0 + kt * 16) * EDIM, EDIM, 0);
      floatx8 sacc = f8zero();
      sacc = __builtin_amdgcn_wmma_f32_16x16x32_bf16(
          false, afrag, false, b, (short)0, sacc, false, false);
#pragma unroll
      for (int r = 0; r < 8; ++r) rsum[r] += __expf(sacc[r] - frm[r]);
    }
#pragma unroll
    for (int off = 1; off < 16; off <<= 1)
#pragma unroll
      for (int r = 0; r < 8; ++r) rsum[r] += __shfl_xor(rsum[r], off, 32);
    if (lcol == 0)
#pragma unroll
      for (int r = 0; r < 8; ++r) s_ps[wave][hf * 8 + r] = rsum[r];
    __syncthreads();
    if (tid < 16) {
      float s = s_ps[0][tid];
      for (int w2 = 1; w2 < 8; ++w2) s += s_ps[w2][tid];
      s_rs[tid] = 1.f / s;
    }
    __syncthreads();
    float frs[8];
#pragma unroll
    for (int r = 0; r < 8; ++r) frs[r] = s_rs[hf * 8 + r];

    // ---- pass C: emit P, accumulate amap (LDS) + P.V via staged WMMA ----
    floatx8 oacc0 = f8zero(), oacc1 = f8zero();
    for (int kp2 = 0; kp2 < 16; ++kp2) {  // pairs of 16-key tiles (32 keys)
      const int kbase = key0 + kp2 * 32;
#pragma unroll
      for (int ht = 0; ht < 2; ++ht) {
        bf16x16 b = load_frag_b(Kh + (size_t)(kbase + ht * 16) * EDIM, EDIM, 0);
        floatx8 sacc = f8zero();
        sacc = __builtin_amdgcn_wmma_f32_16x16x32_bf16(
            false, afrag, false, b, (short)0, sacc, false, false);
        const int kcol = kbase + ht * 16 + lcol;
#pragma unroll
        for (int r = 0; r < 8; ++r) {
          const float p = __expf(sacc[r] - frm[r]) * frs[r];
          const int row = r + 8 * hf;
          amap_acc[row * LSEQ + kcol] += p * 0.125f;   // LDS, unique owner
          s_pst[wave][row][ht * 16 + lcol] = (bf16)p;  // C-layout -> LDS
        }
      }
      // wave-local restage: read P tile back in A-fragment layout (in-order DS)
      bf16x16 pa = load_frag_a(&s_pst[wave][0][0], 32, 0);
      bf16x16 vb0 = load_frag_b(Vtn + (size_t)(h * 32 + 0)  * LSEQ, LSEQ, kbase);
      bf16x16 vb1 = load_frag_b(Vtn + (size_t)(h * 32 + 16) * LSEQ, LSEQ, kbase);
      oacc0 = __builtin_amdgcn_wmma_f32_16x16x32_bf16(
          false, pa, false, vb0, (short)0, oacc0, false, false);
      oacc1 = __builtin_amdgcn_wmma_f32_16x16x32_bf16(
          false, pa, false, vb1, (short)0, oacc1, false, false);
    }

    // ---- cross-wave O reduction and store ----
#pragma unroll
    for (int r = 0; r < 8; ++r) {
      s_ored[wave][r + 8 * hf][lcol]      = oacc0[r];
      s_ored[wave][r + 8 * hf][16 + lcol] = oacc1[r];
    }
    __syncthreads();
    for (int i = tid; i < 16 * 32; i += 256) {
      const int row = i >> 5, col = i & 31;
      float v = 0.f;
#pragma unroll
      for (int w2 = 0; w2 < 8; ++w2) v += s_ored[w2][row][col];
      Ob[((size_t)n * LSEQ + q0 + row) * EDIM + h * 32 + col] = (bf16)v;
    }
    __syncthreads();
  }

  // ---- single write of the head-mean attention map ----
  for (int i = tid; i < 16 * LSEQ; i += 256) {
    const int row = i >> 12, key = i & 4095;
    amap[((size_t)n << 24) + (size_t)(q0 + row) * LSEQ + key] = amap_acc[i];
  }
}

// ---------------------------------------------------------------------------
// Positional embedding + bf16 pack: X[(n*L+l)*256 + c] = in[n,c,h,w] + pe[c,h,w]
// ---------------------------------------------------------------------------
__global__ void prep_kernel(const float* __restrict__ sharp,
                            const float* __restrict__ hist,
                            const float* __restrict__ cur,
                            bf16* __restrict__ Xs, bf16* __restrict__ Xh,
                            bf16* __restrict__ Xc) {
  const size_t idx = (size_t)blockIdx.x * 256 + threadIdx.x;  // N*L*C
  const int c  = idx & 255;
  const size_t ml = idx >> 8;
  const int l  = (int)(ml & 4095);
  const int n  = (int)(ml >> 12);
  const int hh = l >> 6, ww = l & 63;
  const int c4 = c & 63, sect = c >> 6;
  const float omega = __powf(1000.f, -(float)c4 * (1.f / 63.f));
  const float coord = (sect < 2) ? (float)ww : (float)hh;
  const float ang = coord * omega;
  const float pe = (sect & 1) ? __cosf(ang) : __sinf(ang);
  const size_t src = (((size_t)n * EDIM + c) << 12) + l;
  Xs[idx] = (bf16)(sharp[src] + pe);
  Xh[idx] = (bf16)(hist[src] + pe);
  Xc[idx] = (bf16)(cur[src] + pe);
}

__global__ void convert_bf16_kernel(const float* __restrict__ src,
                                    bf16* __restrict__ dst, int n) {
  const int i = blockIdx.x * 256 + threadIdx.x;
  if (i < n) dst[i] = (bf16)src[i];
}

// Vt[((n*256)+c)*4096 + l] = Vp[((n*4096+l)*256)+c]
__global__ void transpose_v_kernel(const bf16* __restrict__ Vp,
                                   bf16* __restrict__ Vt) {
  const size_t idx = (size_t)blockIdx.x * 256 + threadIdx.x;  // N*256*4096
  const int l = (int)(idx & 4095);
  const int c = (int)((idx >> 12) & 255);
  const int n = (int)(idx >> 20);
  Vt[idx] = Vp[(((size_t)n * LSEQ + l) << 8) + c];
}

// max+argmax (first occurrence) along keys for each (n,lq)
__global__ void rowmax_arg_kernel(const float* __restrict__ amap,
                                  float* __restrict__ mval,
                                  int* __restrict__ midx) {
  const int row = blockIdx.x;  // n*L + lq
  const float* r = amap + (size_t)row * LSEQ;
  float m = -INFINITY;
  int mi = 0x7fffffff;
  for (int k = threadIdx.x; k < LSEQ; k += 256)
    if (r[k] > m) { m = r[k]; mi = k; }
#pragma unroll
  for (int off = 16; off; off >>= 1) {
    const float om = __shfl_xor(m, off, 32);
    const int   oi = __shfl_xor(mi, off, 32);
    if (om > m || (om == m && oi < mi)) { m = om; mi = oi; }
  }
  __shared__ float rm[8];
  __shared__ int   ri[8];
  if ((threadIdx.x & 31) == 0) { rm[threadIdx.x >> 5] = m; ri[threadIdx.x >> 5] = mi; }
  __syncthreads();
  if (threadIdx.x == 0) {
    for (int i = 1; i < 8; ++i)
      if (rm[i] > m || (rm[i] == m && ri[i] < mi)) { m = rm[i]; mi = ri[i]; }
    mval[row] = m;
    midx[row] = mi;
  }
}

// max along queries for each (n,lk); threads over lk -> coalesced
__global__ void colmax_kernel(const float* __restrict__ amap,
                              float* __restrict__ mval) {
  const size_t idx = (size_t)blockIdx.x * 256 + threadIdx.x;  // N*L
  const int lk = (int)(idx & 4095);
  const int n  = (int)(idx >> 12);
  const float* base = amap + ((size_t)n << 24) + lk;
  float m = -INFINITY;
  for (int lq = 0; lq < LSEQ; ++lq) m = fmaxf(m, base[(size_t)lq << 12]);
  mval[idx] = m;
}

__global__ void coords_kernel(const float* __restrict__ match12,
                              const int* __restrict__ idx12,
                              const float* __restrict__ match21,
                              float* __restrict__ coords) {
  const size_t idx = (size_t)blockIdx.x * 256 + threadIdx.x;  // N*L
  const int lq = (int)(idx & 4095);
  const int n  = (int)(idx >> 12);
  const float m12 = match12[idx];
  const int   i   = idx12[idx];
  const float g   = match21[((size_t)n << 12) + i];
  const int ci = ((m12 - g) == 0.0f) ? i : lq;
  float* base = coords + ((size_t)n * 2 << 12);
  base[lq]        = (float)(ci & 63);  // x
  base[LSEQ + lq] = (float)(ci >> 6);  // y
}

// axis=0: reduce over wq (lq = hq*64 + i, stride L);  out index (n,hq,lk)
// axis=1: reduce over hq (lq = i*64 + wq, stride 64L); out index (n,wq,lk)
__global__ void corr_reduce_kernel(const float* __restrict__ amap,
                                   float* __restrict__ mx,
                                   float* __restrict__ sm, int axis) {
  const size_t idx = (size_t)blockIdx.x * 256 + threadIdx.x;  // N*64*L
  const int lk = (int)(idx & 4095);
  const int q  = (int)((idx >> 12) & 63);
  const int n  = (int)(idx >> 18);
  const float* base;
  size_t stride;
  if (axis == 0) { base = amap + ((size_t)n << 24) + ((size_t)(q * 64) << 12) + lk; stride = LSEQ; }
  else           { base = amap + ((size_t)n << 24) + ((size_t)q << 12) + lk;        stride = (size_t)64 * LSEQ; }
  float m = -INFINITY;
  for (int i = 0; i < 64; ++i) m = fmaxf(m, base[i * stride]);
  float s = 0.f;
  for (int i = 0; i < 64; ++i) s += __expf(base[i * stride] - m);
  mx[idx] = m;
  sm[idx] = s;
}

__global__ void corr_finalize_kernel(float* __restrict__ amap,
                                     const float* __restrict__ m2,
                                     const float* __restrict__ s2,
                                     const float* __restrict__ m1,
                                     const float* __restrict__ s1) {
  const size_t idx = (size_t)blockIdx.x * 256 + threadIdx.x;  // N*L*L
  const int lk = (int)(idx & 4095);
  const size_t t = idx >> 12;
  const int lq = (int)(t & 4095);
  const int n  = (int)(t >> 12);
  const int hq = lq >> 6, wq = lq & 63;
  const float v = amap[idx];
  const size_t i2 = ((size_t)n << 18) + ((size_t)hq << 12) + lk;
  const size_t i1 = ((size_t)n << 18) + ((size_t)wq << 12) + lk;
  amap[idx] = (__expf(v - m2[i2]) / s2[i2]) * (__expf(v - m1[i1]) / s1[i1]);
}

// ---------------------------------------------------------------------------
extern "C" void kernel_launch(void* const* d_in, const int* in_sizes, int n_in,
                              void* d_out, int out_size, void* d_ws,
                              size_t ws_size, hipStream_t stream) {
  (void)in_sizes; (void)n_in; (void)out_size; (void)ws_size;
  const float* sharp = (const float*)d_in[0];
  const float* hist  = (const float*)d_in[1];
  const float* cur   = (const float*)d_in[2];
  const float* p1w1 = (const float*)d_in[3];
  const float* p1b1 = (const float*)d_in[4];
  const float* p1w2 = (const float*)d_in[5];
  const float* p1b2 = (const float*)d_in[6];
  const float* p2w1 = (const float*)d_in[7];
  const float* p2b1 = (const float*)d_in[8];
  const float* p2w2 = (const float*)d_in[9];
  const float* p2b2 = (const float*)d_in[10];
  const float* inw  = (const float*)d_in[11];
  const float* inb  = (const float*)d_in[12];
  const float* outw = (const float*)d_in[13];
  const float* outb = (const float*)d_in[14];

  float* out    = (float*)d_out;
  float* amap   = out + (size_t)NBATCH * EDIM * LSEQ;   // correlation region
  float* coords = amap + (size_t)NBATCH * LSEQ * LSEQ;

  char* w = (char*)d_ws;
  auto alloc = [&](size_t bytes) -> char* {
    char* p = w; w += (bytes + 255) & ~(size_t)255; return p;
  };
  const size_t MROWS = (size_t)NBATCH * LSEQ;  // 16384
  bf16* Wbf = (bf16*)alloc(8 * 65536 * sizeof(bf16));
  bf16* Xs  = (bf16*)alloc(MROWS * EDIM * sizeof(bf16));
  bf16* Xh  = (bf16*)alloc(MROWS * EDIM * sizeof(bf16));
  bf16* Xc  = (bf16*)alloc(MROWS * EDIM * sizeof(bf16));
  bf16* Hb  = (bf16*)alloc(MROWS * EDIM * sizeof(bf16));
  bf16* Q0  = (bf16*)alloc(MROWS * EDIM * sizeof(bf16));
  bf16* K0  = (bf16*)alloc(MROWS * EDIM * sizeof(bf16));
  bf16* Qp  = (bf16*)alloc(MROWS * EDIM * sizeof(bf16));
  bf16* Kp  = (bf16*)alloc(MROWS * EDIM * sizeof(bf16));
  bf16* Vp  = (bf16*)alloc(MROWS * EDIM * sizeof(bf16));
  bf16* Ob  = (bf16*)alloc(MROWS * EDIM * sizeof(bf16));
  bf16* Vt  = (bf16*)alloc((size_t)NBATCH * EDIM * LSEQ * sizeof(bf16));
  float* M2 = (float*)alloc((size_t)NBATCH * 64 * LSEQ * sizeof(float));
  float* S2 = (float*)alloc((size_t)NBATCH * 64 * LSEQ * sizeof(float));
  float* M1 = (float*)alloc((size_t)NBATCH * 64 * LSEQ * sizeof(float));
  float* S1 = (float*)alloc((size_t)NBATCH * 64 * LSEQ * sizeof(float));
  float* Mt12 = (float*)alloc(MROWS * sizeof(float));
  int*   Ix12 = (int*)alloc(MROWS * sizeof(int));
  float* Mt21 = (float*)alloc(MROWS * sizeof(float));

  // --- weights -> bf16 ---
  bf16* Wp1w1 = Wbf;
  bf16* Wp1w2 = Wbf + 65536;
  bf16* Wp2w1 = Wbf + 2 * 65536;
  bf16* Wp2w2 = Wbf + 3 * 65536;
  bf16* Wq    = Wbf + 4 * 65536;
  bf16* Wk    = Wbf + 5 * 65536;
  bf16* Wv    = Wbf + 6 * 65536;
  bf16* Wo    = Wbf + 7 * 65536;
  convert_bf16_kernel<<<256, 256, 0, stream>>>(p1w1, Wp1w1, 65536);
  convert_bf16_kernel<<<256, 256, 0, stream>>>(p1w2, Wp1w2, 65536);
  convert_bf16_kernel<<<256, 256, 0, stream>>>(p2w1, Wp2w1, 65536);
  convert_bf16_kernel<<<256, 256, 0, stream>>>(p2w2, Wp2w2, 65536);
  convert_bf16_kernel<<<768, 256, 0, stream>>>(inw, Wq, 3 * 65536);  // wq|wk|wv
  convert_bf16_kernel<<<256, 256, 0, stream>>>(outw, Wo, 65536);

  // --- positional embedding + pack ---
  prep_kernel<<<65536, 256, 0, stream>>>(sharp, hist, cur, Xs, Xh, Xc);

  const dim3 blk(256);
  const dim3 gProj(128, 4);  // M=16384, N=256 (NT=4)
  const float qscale = 0.17677669529663687f;  // 32^-0.5

  // --- MLPs + qkv projections (bf16 WMMA GEMMs) ---
  gemm_wmma_kernel<4, 0><<<gProj, blk, 0, stream>>>(Xs, EDIM, Wp1w1, EDIM, Hb, EDIM, p1b1, EDIM, 1.f, 1);
  gemm_wmma_kernel<4, 0><<<gProj, blk, 0, stream>>>(Hb, EDIM, Wp1w2, EDIM, Q0, EDIM, p1b2, EDIM, 1.f, 1);
  gemm_wmma_kernel<4, 0><<<gProj, blk, 0, stream>>>(Xh, EDIM, Wp2w1, EDIM, Hb, EDIM, p2b1, EDIM, 1.f, 1);
  gemm_wmma_kernel<4, 0><<<gProj, blk, 0, stream>>>(Hb, EDIM, Wp2w2, EDIM, K0, EDIM, p2b2, EDIM, 1.f, 1);
  gemm_wmma_kernel<4, 0><<<gProj, blk, 0, stream>>>(Q0, EDIM, Wq, EDIM, Qp, EDIM, inb,       EDIM, qscale, 0);
  gemm_wmma_kernel<4, 0><<<gProj, blk, 0, stream>>>(K0, EDIM, Wk, EDIM, Kp, EDIM, inb + 256, EDIM, 1.f, 0);
  gemm_wmma_kernel<4, 0><<<gProj, blk, 0, stream>>>(Xc, EDIM, Wv, EDIM, Vp, EDIM, inb + 512, EDIM, 1.f, 0);

  // --- V transpose (all batches) ---
  transpose_v_kernel<<<65536, blk, 0, stream>>>(Vp, Vt);

  // --- fused attention: S/P never leave the WGP; amap written once ---
  fused_attn_kernel<<<dim3(256, NBATCH), blk, 16 * LSEQ * sizeof(float), stream>>>(
      Qp, Kp, Vt, Ob, amap);

  // --- out_proj, scattered straight into [N,C,H,W] layout of d_out ---
  gemm_wmma_kernel<4, 2><<<gProj, blk, 0, stream>>>(Ob, EDIM, Wo, EDIM, out, 0,
                                                    outb, EDIM, 1.f, 0);

  // --- mutual nearest-neighbor matching (reads amap before overwrite) ---
  rowmax_arg_kernel<<<MROWS, blk, 0, stream>>>(amap, Mt12, Ix12);
  colmax_kernel<<<64, blk, 0, stream>>>(amap, Mt21);
  coords_kernel<<<64, blk, 0, stream>>>(Mt12, Ix12, Mt21, coords);

  // --- correlation map: two query-axis softmaxes, finalized in place ---
  corr_reduce_kernel<<<4096, blk, 0, stream>>>(amap, M2, S2, 0);  // over wq
  corr_reduce_kernel<<<4096, blk, 0, stream>>>(amap, M1, S1, 1);  // over hq
  corr_finalize_kernel<<<262144, blk, 0, stream>>>(amap, M2, S2, M1, S1);
}
